// DQLinearLoRA_73864847556831
// MI455X (gfx1250) — matrix operand order repaired and checked by reading, not compiled, not run on hardware
//
#include <hip/hip_runtime.h>
#include <hip/hip_bf16.h>

typedef __attribute__((ext_vector_type(16))) __bf16 v16bf;
typedef __attribute__((ext_vector_type(8)))  __bf16 v8bf;
typedef __attribute__((ext_vector_type(4)))  __bf16 v4bf;
typedef __attribute__((ext_vector_type(8)))  float  v8f;

// ---- gfx1250 async-to-LDS copy (16B per lane), tracked by ASYNCcnt --------
__device__ __forceinline__ void async_ld16(unsigned lds_off, const __bf16* gaddr) {
    asm volatile("global_load_async_to_lds_b128 %0, %1, off"
                 :: "v"(lds_off), "v"(gaddr)
                 : "memory");
}
#define WAIT_ASYNC(n) asm volatile("s_wait_asynccnt " #n ::: "memory")

// ---------------------------------------------------------------------------
// Kernel 1: W_eff[n,k] = dequant_nf4(weight)[n,k] + (lora_B @ lora_A)[n,k]
//           stored bf16. 64x64 tile / 256-thread block; lora tiles in LDS.
// ---------------------------------------------------------------------------
__global__ void __launch_bounds__(256)
prep_weff_kernel(const float* __restrict__ weight,
                 const float* __restrict__ lora_A,
                 const float* __restrict__ lora_B,
                 const float* __restrict__ max_val,
                 const float* __restrict__ lut,
                 __hip_bfloat16* __restrict__ weff,
                 int Kdim, int rank)
{
    __shared__ float sLut[16];
    __shared__ float sB[64 * 64];   // lora_B tile: [row_in_tile][rank]
    __shared__ float sA[64 * 64];   // lora_A tile: [rank][col_in_tile]
    const int tid = threadIdx.x;
    if (tid < 16) sLut[tid] = lut[tid];
    const int r0 = blockIdx.y * 64;
    const int c0 = blockIdx.x * 64;
    for (int i = tid; i < 64 * 64; i += 256) {
        sB[i] = lora_B[(size_t)(r0 + (i >> 6)) * rank + (i & 63)];
        sA[i] = lora_A[(size_t)(i >> 6) * Kdim + c0 + (i & 63)];
    }
    __syncthreads();

    const int c = tid & 63;
    #pragma unroll 1
    for (int j = 0; j < 16; ++j) {
        const int r = (tid >> 6) + 4 * j;                 // 0..63
        const size_t f = (size_t)(r0 + r) * Kdim + (c0 + c);
        const float scale = max_val[f >> 6];              // block = 64
        const float w = weight[f] / scale;
        float best = 3.4e38f, q = 0.f;                    // first-min == jnp.argmin
        #pragma unroll
        for (int i = 0; i < 16; ++i) {
            const float d = fabsf(w - sLut[i]);
            if (d < best) { best = d; q = sLut[i]; }
        }
        float acc = q * scale;
        #pragma unroll
        for (int k = 0; k < 64; ++k)
            acc = fmaf(sB[r * 64 + k], sA[k * 64 + c], acc);
        weff[f] = __float2bfloat16(acc);
    }
}

// ---------------------------------------------------------------------------
// Kernel 2: x (f32) -> bf16, streaming.
// ---------------------------------------------------------------------------
__global__ void __launch_bounds__(256)
cvt_x_kernel(const float* __restrict__ x, __hip_bfloat16* __restrict__ xb, long n)
{
    const long i = ((long)blockIdx.x * blockDim.x + threadIdx.x) * 4;
    if (i < n) {
        const float4 v = *(const float4*)(x + i);
        v4bf o;
        o[0] = (__bf16)v.x; o[1] = (__bf16)v.y;
        o[2] = (__bf16)v.z; o[3] = (__bf16)v.w;
        *(v4bf*)((__bf16*)xb + i) = o;
    }
}

// ---------------------------------------------------------------------------
// Kernel 3: out[M,N] = Xb[M,K] @ Wb[N,K]^T  (bf16 WMMA, f32 accumulate).
// 128 threads = 4 wave32; block tile 256(M) x 64(N); wave tile 64x64 = 4x4
// v_wmma_f32_16x16x32_bf16 tiles.  The shared 64-row W_eff slab is staged in
// LDS via GLOBAL_LOAD_ASYNC_TO_LDS_B128 (ASYNCcnt, double-buffered, 64-wide
// K stages); A fragments load direct from global (wave-private rows).
// LDS B row stride = 72 elems (144B): bank = 4*(9r mod 16) -> conflict-free.
// ---------------------------------------------------------------------------
#define KSTAGE   64
#define BROWSTRD 72                       // bf16 elems per LDS row (64 + 8 pad)

__global__ void __launch_bounds__(128)
gemm_wmma_kernel(const __hip_bfloat16* __restrict__ Xb,
                 const __hip_bfloat16* __restrict__ Wb,
                 float* __restrict__ out,
                 int M, int N, int K)
{
    __shared__ __align__(16) __bf16 sB[2][64 * BROWSTRD];

    const int tid  = threadIdx.x;
    const int lane = tid & 31;
    const int wave = tid >> 5;
    const int hf   = lane >> 4;        // K-half selector
    const int ln   = lane & 15;        // A row / B,D column

    const int m0 = blockIdx.y * 256 + wave * 64;
    const int n0 = blockIdx.x * 64;

    const __bf16* Xp = (const __bf16*)Xb;
    const __bf16* Wp = (const __bf16*)Wb;

    // per-lane A base pointers (row-major x, K contiguous)
    const __bf16* ap[4];
    #pragma unroll
    for (int i = 0; i < 4; ++i)
        ap[i] = Xp + (size_t)(m0 + i * 16 + ln) * K + hf * 8;

    const unsigned sB_base = (unsigned)(size_t)&sB[0][0];
    const unsigned bufBytes = 64u * BROWSTRD * 2u;

    // cooperative async fill of one 64x64 B slab: 512 16B chunks / 128 thr
    auto fill = [&](int buf, int kk) {
        #pragma unroll
        for (int c = 0; c < 4; ++c) {
            const int chunk = tid + c * 128;          // 0..511
            const int row   = chunk >> 3;             // 0..63
            const int o16   = chunk & 7;              // 16B chunk in row
            const unsigned lo = sB_base + (unsigned)buf * bufBytes
                              + (unsigned)row * (BROWSTRD * 2u) + (unsigned)o16 * 16u;
            const __bf16* g = Wp + (size_t)(n0 + row) * K + kk + o16 * 8;
            async_ld16(lo, g);
        }
    };

    v8f acc[4][4] = {};

    const int steps = K / KSTAGE;
    fill(0, 0);                                       // prologue

    for (int s = 0; s < steps; ++s) {
        const int cur = s & 1;
        const int kk  = s * KSTAGE;
        if (s + 1 < steps) {
            fill(cur ^ 1, kk + KSTAGE);               // prefetch next stage
            WAIT_ASYNC(0x4);                          // own cur-stage fills done
        } else {
            WAIT_ASYNC(0x0);
        }
        __syncthreads();                              // all waves' fills visible

        const __bf16* sBc = &sB[cur][0];
        #pragma unroll
        for (int s2 = 0; s2 < 2; ++s2) {              // two 32-wide WMMA K-steps
            const int kl = s2 * 32;
            v16bf a[4], b[4];
            #pragma unroll
            for (int j = 0; j < 4; ++j) {             // B frags from LDS
                const __bf16* p = sBc + (size_t)(j * 16 + ln) * BROWSTRD + hf * 8 + kl;
                const v8bf lo = *(const v8bf*)(p);
                const v8bf hi = *(const v8bf*)(p + 16);
                #pragma unroll
                for (int t = 0; t < 8; ++t) { b[j][t] = lo[t]; b[j][t + 8] = hi[t]; }
            }
            #pragma unroll
            for (int i = 0; i < 4; ++i) {             // A frags direct global
                const v8bf lo = *(const v8bf*)(ap[i] + kk + kl);
                const v8bf hi = *(const v8bf*)(ap[i] + kk + kl + 16);
                #pragma unroll
                for (int t = 0; t < 8; ++t) { a[i][t] = lo[t]; a[i][t + 8] = hi[t]; }
            }
            #pragma unroll
            for (int i = 0; i < 4; ++i)
                #pragma unroll
                for (int j = 0; j < 4; ++j)
                    acc[i][j] = __builtin_amdgcn_wmma_f32_16x16x32_bf16(
                        false, a[i], false, b[j], (short)0, acc[i][j], false, false);
        }
        __syncthreads();                              // reads done before overwrite
    }

    // D layout: lane = column (ln), VGPR t = row; lanes>=16 hold rows 8..15
    #pragma unroll
    for (int i = 0; i < 4; ++i) {
        #pragma unroll
        for (int j = 0; j < 4; ++j) {
            float* o = out + (size_t)(m0 + i * 16 + hf * 8) * N + (n0 + j * 16 + ln);
            #pragma unroll
            for (int t = 0; t < 8; ++t)
                o[(size_t)t * N] = acc[i][j][t];
        }
    }
}

// ---------------------------------------------------------------------------
extern "C" void kernel_launch(void* const* d_in, const int* in_sizes, int n_in,
                              void* d_out, int out_size, void* d_ws, size_t ws_size,
                              hipStream_t stream)
{
    const float* x       = (const float*)d_in[0];   // [8192, 4096]
    const float* weight  = (const float*)d_in[1];   // [4096, 4096]
    const float* lora_A  = (const float*)d_in[2];   // [64, 4096]
    const float* lora_B  = (const float*)d_in[3];   // [4096, 64]
    const float* max_val = (const float*)d_in[4];   // [262144]
    const float* lut     = (const float*)d_in[5];   // [16]
    float* out = (float*)d_out;

    const int Kd = 4096;   // in_features
    const int Nd = 4096;   // out_features
    const int Md = 8192;   // tokens
    const int Rk = 64;     // reduced rank

    // workspace: [W_eff bf16: N*K*2 = 32MB][x bf16: M*K*2 = 64MB]
    __hip_bfloat16* weff = (__hip_bfloat16*)d_ws;
    __hip_bfloat16* xb   = (__hip_bfloat16*)((char*)d_ws + (size_t)Nd * Kd * 2);

    prep_weff_kernel<<<dim3(Kd / 64, Nd / 64), 256, 0, stream>>>(
        weight, lora_A, lora_B, max_val, lut, weff, Kd, Rk);

    const long nx = (long)Md * Kd;
    cvt_x_kernel<<<(unsigned)((nx / 4 + 255) / 256), 256, 0, stream>>>(x, xb, nx);

    gemm_wmma_kernel<<<dim3(Nd / 64, Md / 256), 128, 0, stream>>>(
        xb, weff, out, Md, Nd, Kd);
}
